// MvHeat_DET_40956808135072
// MI455X (gfx1250) — matrix operand order repaired and checked by reading, not compiled, not run on hardware
//
#include <hip/hip_runtime.h>
#include <hip/hip_bf16.h>
#include <math.h>

typedef __attribute__((ext_vector_type(16))) _Float16 v16h;
typedef __attribute__((ext_vector_type(8)))  float    v8f;

#define PI_F 3.14159265358979323846f
static const long long BIGN = 32LL * 64 * 64 * 128;   // 16,777,216 elems
static const long long PB   = 64LL * 64 * 128;        // 524,288 per-batch

// ======================================================================
// Generic batched WMMA GEMM:  C[bi](MxN) = alpha*(A[bi] @ B[bi]) [+C] [+bias]
// A: MxK row-major (f32 -> f16 in-register)
// B: KxN row-major, or (TRANSB) NxK row-major; entire K x 32n panel staged
//    to LDS as f16 (K-major) up-front, single barrier, then barrier-free
//    compute. Each wave computes a 16x32 tile (2 WMMAs share one A frag).
// Gating: skip batch bi unless sel[bi/selDiv]==branch (sel may be null).
// Requirements: M % 64 == 0, N % 32 == 0, KK in {64,128}.
// ======================================================================
template <int TRANSB, int KK>
__global__ __launch_bounds__(128) void gemm_wmma(
    float* __restrict__ C, const float* __restrict__ A, const float* __restrict__ B,
    int M, int N,
    long long cStride, long long aStride, long long bStride,
    float alpha, int beta, const float* __restrict__ bias,
    const int* __restrict__ sel, int selDiv, int branch)
{
  // [step s][n 0..31][k 0..31] f16, 2KB per step
  __shared__ alignas(32) _Float16 lb[(KK / 32) * 32 * 32];
  int bi = blockIdx.z;
  if (sel && sel[bi / selDiv] != branch) return;   // block-uniform
  const float* Ab = A + (long long)bi * aStride;
  const float* Bb = B + (long long)bi * bStride;
  float*       Cb = C + (long long)bi * cStride;

  int tid  = threadIdx.x;
  int wave = tid >> 5;
  int lane = tid & 31;
  int tm = blockIdx.y * 64 + wave * 16;
  int tn = blockIdx.x * 32;
  int l = lane & 15;
  int g = lane >> 4;
  int am = tm + l;            // A row held by this lane

  // ---- stage ALL B tiles (KK x 32n) into LDS, coalesced float4 loads ----
  #pragma unroll
  for (int s = 0; s < KK / 32; ++s) {
    int k0 = s * 32;
    if (TRANSB) {
      int n  = tid >> 2;                 // 0..31
      int k8 = (tid & 3) << 3;           // 0,8,16,24
      const float* p = Bb + (long long)(tn + n) * KK + (k0 + k8);
      float4 v0 = *(const float4*)p;
      float4 v1 = *(const float4*)(p + 4);
      union { _Float16 h[8]; uint4 u; } pk;
      pk.h[0] = (_Float16)v0.x; pk.h[1] = (_Float16)v0.y;
      pk.h[2] = (_Float16)v0.z; pk.h[3] = (_Float16)v0.w;
      pk.h[4] = (_Float16)v1.x; pk.h[5] = (_Float16)v1.y;
      pk.h[6] = (_Float16)v1.z; pk.h[7] = (_Float16)v1.w;
      *(uint4*)&lb[(s * 32 + n) * 32 + k8] = pk.u;
    } else {
      int k  = tid >> 2;                 // 0..31
      int n8 = (tid & 3) << 3;           // 0,8,16,24
      const float* p = Bb + (long long)(k0 + k) * N + tn + n8;
      float4 v0 = *(const float4*)p;
      float4 v1 = *(const float4*)(p + 4);
      lb[(s * 32 + n8 + 0) * 32 + k] = (_Float16)v0.x;
      lb[(s * 32 + n8 + 1) * 32 + k] = (_Float16)v0.y;
      lb[(s * 32 + n8 + 2) * 32 + k] = (_Float16)v0.z;
      lb[(s * 32 + n8 + 3) * 32 + k] = (_Float16)v0.w;
      lb[(s * 32 + n8 + 4) * 32 + k] = (_Float16)v1.x;
      lb[(s * 32 + n8 + 5) * 32 + k] = (_Float16)v1.y;
      lb[(s * 32 + n8 + 6) * 32 + k] = (_Float16)v1.z;
      lb[(s * 32 + n8 + 7) * 32 + k] = (_Float16)v1.w;
    }
  }
  __syncthreads();   // the only barrier

  v8f acc0 = {0.f,0.f,0.f,0.f,0.f,0.f,0.f,0.f};
  v8f acc1 = {0.f,0.f,0.f,0.f,0.f,0.f,0.f,0.f};

  #pragma unroll
  for (int s = 0; s < KK / 32; ++s) {
    int k0 = s * 32;
    // ---- A fragment: contiguous k = 16*g + element (matches B) ----
    v16h a;
    #pragma unroll
    for (int v = 0; v < 8; ++v) {
      int kk = k0 + (g << 4) + (v << 1);
      const float* ap = Ab + (long long)am * KK + kk;
      a[2 * v]     = (_Float16)ap[0];
      a[2 * v + 1] = (_Float16)ap[1];
    }
    // ---- B fragments: contiguous 32B from LDS (2x ds_load_b128 each) ----
    v16h b0 = *(const v16h*)&lb[(s * 32 + l) * 32 + (g << 4)];
    v16h b1 = *(const v16h*)&lb[(s * 32 + 16 + l) * 32 + (g << 4)];

    acc0 = __builtin_amdgcn_wmma_f32_16x16x32_f16(false, a, false, b0, (short)0,
                                                  acc0, false, false);
    acc1 = __builtin_amdgcn_wmma_f32_16x16x32_f16(false, a, false, b1, (short)0,
                                                  acc1, false, false);
  }

  float bs0 = bias ? bias[tn + l] : 0.0f;
  float bs1 = bias ? bias[tn + 16 + l] : 0.0f;
  #pragma unroll
  for (int v = 0; v < 8; ++v) {
    long long row = (long long)(tm + v + 8 * g) * N;
    long long c0 = row + tn + l;
    long long c1 = row + tn + 16 + l;
    float val0 = alpha * acc0[v] + bs0;
    float val1 = alpha * acc1[v] + bs1;
    if (beta) { val0 += Cb[c0]; val1 += Cb[c1]; }
    Cb[c0] = val0;
    Cb[c1] = val1;
  }
}

// ======================================================================
// Small matrices: DCT-II (wn, wnT) and DFT cos/sin (64 and 128)
// ======================================================================
__global__ void init_mats(float* wn, float* wnT, float* f64c, float* f64s,
                          float* f128c, float* f128s)
{
  int idx = blockIdx.x * 256 + threadIdx.x;
  if (idx < 4096) {
    int n = idx >> 6, h = idx & 63;
    float v = cosf((float)n * ((h + 0.5f) / 64.0f) * PI_F) * sqrtf(2.0f / 64.0f);
    if (n == 0) v *= 0.70710678118654752440f;
    wn[n * 64 + h] = v;
    wnT[h * 64 + n] = v;
  } else if (idx < 8192) {
    int i = idx - 4096;
    int k = i >> 6, n = i & 63;
    int t = (k * n) & 63;
    float ang = (float)t * (2.0f * PI_F / 64.0f);
    f64c[i] = cosf(ang);
    f64s[i] = sinf(ang);
  } else if (idx < 8192 + 16384) {
    int i = idx - 8192;
    int k = i >> 7, n = i & 127;
    int t = (k * n) & 127;
    float ang = (float)t * (2.0f * PI_F / 128.0f);
    f128c[i] = cosf(ang);
    f128s[i] = sinf(ang);
  }
}

// depthwise 3x3 SAME conv, NCHW input -> BHWC f32 output
__global__ void dwconv_bhwc(const float* __restrict__ x, const float* __restrict__ wgt,
                            const float* __restrict__ bias, float* __restrict__ xc)
{
  long long idx = blockIdx.x * 256LL + threadIdx.x;   // (b,h,w,c)
  int c = (int)(idx & 127);
  int w = (int)((idx >> 7) & 63);
  int h = (int)((idx >> 13) & 63);
  int b = (int)(idx >> 19);
  float s = bias[c];
  const float* xb = x + ((long long)b * 128 + c) * 4096;
  #pragma unroll
  for (int dy = 0; dy < 3; ++dy) {
    int hy = h + dy - 1;
    if ((unsigned)hy >= 64u) continue;
    #pragma unroll
    for (int dx = 0; dx < 3; ++dx) {
      int wx = w + dx - 1;
      if ((unsigned)wx >= 64u) continue;
      s += xb[hy * 64 + wx] * wgt[c * 9 + dy * 3 + dx];
    }
  }
  xc[idx] = s;
}

__global__ void silu_ip(float* z)
{
  long long i = blockIdx.x * 256LL + threadIdx.x;
  float v = z[i];
  z[i] = v / (1.0f + expf(-v));
}

// h_mean[b][c] = mean over 4096 positions of x_[b,:,:,c]
__global__ void rowmean(const float* __restrict__ x_, float* __restrict__ hm)
{
  int b = blockIdx.x, c = threadIdx.x;
  const float* p = x_ + (long long)b * PB + c;
  float s = 0.f;
  for (int i = 0; i < 4096; ++i) s += p[(long long)i * 128];
  hm[b * 128 + c] = s * (1.0f / 4096.0f);
}

// Router MLP: h@p1 -> lrelu -> BN -> @p2 -> lrelu -> BN -> @p3 +gumbel -> argmax
__global__ __launch_bounds__(256) void mlp_moe(
    const float* __restrict__ hmean,
    const float* __restrict__ p1w, const float* __restrict__ p1b,
    const float* __restrict__ bn1g, const float* __restrict__ bn1b,
    const float* __restrict__ p2w, const float* __restrict__ p2b,
    const float* __restrict__ bn2g, const float* __restrict__ bn2b,
    const float* __restrict__ p3w, const float* __restrict__ p3b,
    const float* __restrict__ gumbel, int* __restrict__ sel)
{
  __shared__ float hm[32 * 128];
  __shared__ float h1[32 * 256];
  __shared__ float h2[32 * 32];
  __shared__ float sc[256], sh[256];
  int tid = threadIdx.x;
  for (int i = tid; i < 4096; i += 256) hm[i] = hmean[i];
  __syncthreads();
  for (int i = tid; i < 8192; i += 256) {
    int b = i >> 8, o = i & 255;
    float s = p1b[o];
    for (int c = 0; c < 128; ++c) s += hm[b * 128 + c] * p1w[o * 128 + c];
    s = s > 0.f ? s : 0.1f * s;
    h1[i] = s;
  }
  __syncthreads();
  if (tid < 256) {
    float mu = 0.f, m2 = 0.f;
    for (int b = 0; b < 32; ++b) mu += h1[b * 256 + tid];
    mu *= (1.0f / 32.0f);
    for (int b = 0; b < 32; ++b) { float v = h1[b * 256 + tid] - mu; m2 += v * v; }
    m2 *= (1.0f / 32.0f);
    float s = bn1g[tid] * rsqrtf(m2 + 1e-5f);
    sc[tid] = s;
    sh[tid] = bn1b[tid] - mu * s;
  }
  __syncthreads();
  for (int i = tid; i < 8192; i += 256) { int o = i & 255; h1[i] = h1[i] * sc[o] + sh[o]; }
  __syncthreads();
  for (int i = tid; i < 1024; i += 256) {
    int b = i >> 5, o = i & 31;
    float s = p2b[o];
    for (int c = 0; c < 256; ++c) s += h1[b * 256 + c] * p2w[o * 256 + c];
    s = s > 0.f ? s : 0.1f * s;
    h2[i] = s;
  }
  __syncthreads();
  if (tid < 32) {
    float mu = 0.f, m2 = 0.f;
    for (int b = 0; b < 32; ++b) mu += h2[b * 32 + tid];
    mu *= (1.0f / 32.0f);
    for (int b = 0; b < 32; ++b) { float v = h2[b * 32 + tid] - mu; m2 += v * v; }
    m2 *= (1.0f / 32.0f);
    float s = bn2g[tid] * rsqrtf(m2 + 1e-5f);
    sc[tid] = s;
    sh[tid] = bn2b[tid] - mu * s;
  }
  __syncthreads();
  for (int i = tid; i < 1024; i += 256) { int o = i & 31; h2[i] = h2[i] * sc[o] + sh[o]; }
  __syncthreads();
  if (tid < 32) {
    int b = tid, bi = 0;
    float best = -3.4e38f;
    for (int r = 0; r < 3; ++r) {
      float s = p3b[r];
      for (int j = 0; j < 32; ++j) s += h2[b * 32 + j] * p3w[r * 32 + j];
      s += gumbel[b * 3 + r];
      if (s > best) { best = s; bi = r; }
    }
    sel[b] = bi;
  }
}

// webuf currently holds tok-linear output; replace with exp(-(n^2+m^2)*relu(k))
__global__ void build_we(float* we)
{
  int idx = blockIdx.x * 256 + threadIdx.x;  // (n,m,c), 524288 total
  int m = (idx >> 7) & 63;
  int n = idx >> 13;
  float nn = (float)n * (PI_F / 64.0f);
  float mm = (float)m * (PI_F / 64.0f);
  float k = we[idx];
  k = k > 0.f ? k : 0.f;
  we[idx] = expf(-(nn * nn + mm * mm) * k);
}

// multiply spectral buffers by we (gated per-batch)
__global__ void scale_we(float* p, float* q, const float* __restrict__ we,
                         const int* __restrict__ sel, int branch)
{
  long long idx = blockIdx.x * 256LL + threadIdx.x;
  int b = (int)(idx >> 19);
  if (sel[b] != branch) return;
  float w = we[idx & (PB - 1)];
  p[idx] *= w;
  if (q) q[idx] *= w;
}

// fused Haar forward along H and W + we multiply (branch 2)
__global__ void haar_fwd(const float* __restrict__ x_, const float* __restrict__ we,
                         const int* __restrict__ sel, float* __restrict__ out)
{
  long long idx = blockIdx.x * 256LL + threadIdx.x;
  int b = (int)(idx >> 19);
  if (sel[b] != 2) return;
  int c = (int)(idx & 127);
  int m = (int)((idx >> 7) & 63);
  int n = (int)((idx >> 13) & 63);
  float shs = (n < 32) ? 1.f : -1.f;
  float sws = (m < 32) ? 1.f : -1.f;
  int r0 = (n & 31) * 2, c0 = (m & 31) * 2;
  const float* base = x_ + ((long long)b << 19);
  float x00 = base[(long long)((r0 * 64 + c0) << 7) + c];
  float x01 = base[(long long)((r0 * 64 + c0 + 1) << 7) + c];
  float x10 = base[(long long)(((r0 + 1) * 64 + c0) << 7) + c];
  float x11 = base[(long long)(((r0 + 1) * 64 + c0 + 1) << 7) + c];
  float v = 0.25f * (x00 + sws * x01 + shs * (x10 + sws * x11));
  out[idx] = v * we[idx & (PB - 1)];
}

// fused inverse Haar (H then W) writing xo (branch 2)
__global__ void haar_inv(const float* __restrict__ t, const int* __restrict__ sel,
                         float* __restrict__ xo)
{
  long long idx = blockIdx.x * 256LL + threadIdx.x;
  int b = (int)(idx >> 19);
  if (sel[b] != 2) return;
  int c = (int)(idx & 127);
  int w = (int)((idx >> 7) & 63);
  int h = (int)((idx >> 13) & 63);
  int i = h >> 1, j = w >> 1;
  float sp = (h & 1) ? -1.f : 1.f;
  float sq = (w & 1) ? -1.f : 1.f;
  const float* base = t + ((long long)b << 19);
  float t00 = base[(long long)((i * 64 + j) << 7) + c];
  float t01 = base[(long long)((i * 64 + j + 32) << 7) + c];
  float t10 = base[(long long)(((i + 32) * 64 + j) << 7) + c];
  float t11 = base[(long long)(((i + 32) * 64 + j + 32) << 7) + c];
  xo[idx] = t00 + sq * t01 + sp * (t10 + sq * t11);
}

// LayerNorm(channel) * silu(z); one wave per spatial position
__global__ __launch_bounds__(256) void ln_silu(
    const float* __restrict__ xo, const float* __restrict__ zs,
    const float* __restrict__ g, const float* __restrict__ bta,
    float* __restrict__ out)
{
  int wave = threadIdx.x >> 5, lane = threadIdx.x & 31;
  long long pos = blockIdx.x * 8LL + wave;
  const float* p = xo + pos * 128 + lane * 4;
  float4 v = *(const float4*)p;
  float s = v.x + v.y + v.z + v.w;
  for (int off = 16; off; off >>= 1) s += __shfl_xor(s, off, 32);
  float mu = s * (1.0f / 128.0f);
  float dx = v.x - mu, dy = v.y - mu, dz = v.z - mu, dw = v.w - mu;
  float q = dx * dx + dy * dy + dz * dz + dw * dw;
  for (int off = 16; off; off >>= 1) q += __shfl_xor(q, off, 32);
  float rstd = rsqrtf(q * (1.0f / 128.0f) + 1e-5f);
  float4 z  = *(const float4*)(zs + pos * 128 + lane * 4);
  float4 gg = *(const float4*)(g + lane * 4);
  float4 bb = *(const float4*)(bta + lane * 4);
  float4 o;
  o.x = (dx * rstd * gg.x + bb.x) * z.x;
  o.y = (dy * rstd * gg.y + bb.y) * z.y;
  o.z = (dz * rstd * gg.z + bb.z) * z.z;
  o.w = (dw * rstd * gg.w + bb.w) * z.w;
  *(float4*)(out + pos * 128 + lane * 4) = o;
}

// (b,h,w,o) -> (b,o,h,w)
__global__ void transpose_out(const float* __restrict__ t, float* __restrict__ out)
{
  long long idx = blockIdx.x * 256LL + threadIdx.x;   // NCHW flat
  int w = (int)(idx & 63);
  int h = (int)((idx >> 6) & 63);
  int o = (int)((idx >> 12) & 127);
  int b = (int)(idx >> 19);
  out[idx] = t[(((long long)b * 64 + h) * 64 + w) * 128 + o];
}

// ======================================================================
static void gemm(hipStream_t s, float* C, const float* A, const float* B,
                 int M, int N, int K, long long cs, long long as, long long bs,
                 int batches, float alpha, int beta, const float* bias, int transB,
                 const int* sel, int selDiv, int branch)
{
  dim3 grid(N / 32, M / 64, batches), blk(128);
  if (transB) {
    if (K == 128)
      gemm_wmma<1, 128><<<grid, blk, 0, s>>>(C, A, B, M, N, cs, as, bs, alpha, beta,
                                             bias, sel, selDiv, branch);
    else
      gemm_wmma<1, 64><<<grid, blk, 0, s>>>(C, A, B, M, N, cs, as, bs, alpha, beta,
                                            bias, sel, selDiv, branch);
  } else {
    if (K == 128)
      gemm_wmma<0, 128><<<grid, blk, 0, s>>>(C, A, B, M, N, cs, as, bs, alpha, beta,
                                             bias, sel, selDiv, branch);
    else
      gemm_wmma<0, 64><<<grid, blk, 0, s>>>(C, A, B, M, N, cs, as, bs, alpha, beta,
                                            bias, sel, selDiv, branch);
  }
}

extern "C" void kernel_launch(void* const* d_in, const int* in_sizes, int n_in,
                              void* d_out, int out_size, void* d_ws, size_t ws_size,
                              hipStream_t stream)
{
  (void)in_sizes; (void)n_in; (void)out_size; (void)ws_size;
  const float* x     = (const float*)d_in[0];
  const float* fe    = (const float*)d_in[1];
  const float* dw_w  = (const float*)d_in[2];
  const float* dw_b  = (const float*)d_in[3];
  const float* lin_w = (const float*)d_in[4];
  const float* lin_b = (const float*)d_in[5];
  const float* p1w   = (const float*)d_in[6];
  const float* p1b   = (const float*)d_in[7];
  const float* bn1g  = (const float*)d_in[8];
  const float* bn1b  = (const float*)d_in[9];
  const float* p2w   = (const float*)d_in[10];
  const float* p2b   = (const float*)d_in[11];
  const float* bn2g  = (const float*)d_in[12];
  const float* bn2b  = (const float*)d_in[13];
  const float* p3w   = (const float*)d_in[14];
  const float* p3b   = (const float*)d_in[15];
  const float* tokw  = (const float*)d_in[16];
  const float* tokb  = (const float*)d_in[17];
  const float* ong   = (const float*)d_in[18];
  const float* onb   = (const float*)d_in[19];
  const float* olw   = (const float*)d_in[20];
  const float* olb   = (const float*)d_in[21];
  const float* gum   = (const float*)d_in[22];

  char* wsb = (char*)d_ws;
  size_t off = 0;
  auto take = [&](size_t bytes) -> void* {
    void* p = wsb + off;
    off += (bytes + 255) & ~(size_t)255;
    return p;
  };
  const size_t BIGB = (size_t)BIGN * 4;
  float* xc    = (float*)take(BIGB);   // conv out, reused as P1
  float* x_    = (float*)take(BIGB);   // x_ branch input; reused as xn after LN
  float* zs    = (float*)take(BIGB);   // silu(z)
  float* xo    = (float*)take(BIGB);   // branch-merged output
  float* P0    = (float*)take(BIGB);   // transform temp / final GEMM temp
  float* Q0    = (float*)take(BIGB);   // FFT imag temp
  float* wn    = (float*)take(4096 * 4);
  float* wnT   = (float*)take(4096 * 4);
  float* f64c  = (float*)take(4096 * 4);
  float* f64s  = (float*)take(4096 * 4);
  float* f128c = (float*)take(16384 * 4);
  float* f128s = (float*)take(16384 * 4);
  float* webuf = (float*)take((size_t)PB * 4);
  float* hmean = (float*)take(4096 * 4);
  int*   sel   = (int*)take(32 * 4);
  float* P1 = xc;              // real ping-pong (xc dead after lin GEMMs)
  float* Q1 = (float*)d_out;   // imag ping-pong (d_out fully rewritten at end)

  const long long NB = BIGN / 256;     // 65536 blocks for elementwise

  // 1) small transform matrices
  init_mats<<<96, 256, 0, stream>>>(wn, wnT, f64c, f64s, f128c, f128s);
  // 2) depthwise conv -> BHWC
  dwconv_bhwc<<<(unsigned)NB, 256, 0, stream>>>(x, dw_w, dw_b, xc);
  // 3) x_  = xc @ lin_w[0:128]^T + b   (M=131072,N=128,K=128)
  gemm(stream, x_, xc, lin_w, 131072, 128, 128, 0, 0, 0, 1, 1.f, 0, lin_b, 1,
       nullptr, 1, 0);
  // 4) z = xc @ lin_w[128:256]^T + b ; silu in place
  gemm(stream, zs, xc, lin_w + 128 * 128, 131072, 128, 128, 0, 0, 0, 1, 1.f, 0,
       lin_b + 128, 1, nullptr, 1, 0);
  silu_ip<<<(unsigned)NB, 256, 0, stream>>>(zs);
  // 5) router
  rowmean<<<32, 128, 0, stream>>>(x_, hmean);
  mlp_moe<<<1, 256, 0, stream>>>(hmean, p1w, p1b, bn1g, bn1b, p2w, p2b, bn2g, bn2b,
                                 p3w, p3b, gum, sel);
  // 6) we = decay ** relu(freq_embed @ tok_w^T + tok_b)
  gemm(stream, webuf, fe, tokw, 4096, 128, 128, 0, 0, 0, 1, 1.f, 0, tokb, 1,
       nullptr, 1, 0);
  build_we<<<2048, 256, 0, stream>>>(webuf);

  // ---------------- DCT branch (sel==0) ----------------
  gemm(stream, P0, wn, x_, 64, 8192, 64, PB, 0, PB, 32, 1.f, 0, nullptr, 0, sel, 1, 0);
  gemm(stream, P1, wn, P0, 64, 128, 64, 8192, 0, 8192, 2048, 1.f, 0, nullptr, 0, sel, 64, 0);
  scale_we<<<(unsigned)NB, 256, 0, stream>>>(P1, nullptr, webuf, sel, 0);
  gemm(stream, P0, wnT, P1, 64, 8192, 64, PB, 0, PB, 32, 1.f, 0, nullptr, 0, sel, 1, 0);
  gemm(stream, xo, wnT, P0, 64, 128, 64, 8192, 0, 8192, 2048, 1.f, 0, nullptr, 0, sel, 64, 0);

  // ---------------- FFT branch (sel==1): 3D DFT over (H,W,C) ----------------
  // forward H
  gemm(stream, P0, f64c, x_, 64, 8192, 64, PB, 0, PB, 32,  1.f, 0, nullptr, 0, sel, 1, 1);
  gemm(stream, Q0, f64s, x_, 64, 8192, 64, PB, 0, PB, 32, -1.f, 0, nullptr, 0, sel, 1, 1);
  // forward W:  R2 = C@R + S@I ; I2 = C@I - S@R
  gemm(stream, P1, f64c, P0, 64, 128, 64, 8192, 0, 8192, 2048,  1.f, 0, nullptr, 0, sel, 64, 1);
  gemm(stream, P1, f64s, Q0, 64, 128, 64, 8192, 0, 8192, 2048,  1.f, 1, nullptr, 0, sel, 64, 1);
  gemm(stream, Q1, f64c, Q0, 64, 128, 64, 8192, 0, 8192, 2048,  1.f, 0, nullptr, 0, sel, 64, 1);
  gemm(stream, Q1, f64s, P0, 64, 128, 64, 8192, 0, 8192, 2048, -1.f, 1, nullptr, 0, sel, 64, 1);
  // forward C:  R3 = R2@Fc + I2@Fs ; I3 = I2@Fc - R2@Fs   (F symmetric)
  gemm(stream, P0, P1, f128c, 4096, 128, 128, PB, PB, 0, 32,  1.f, 0, nullptr, 0, sel, 1, 1);
  gemm(stream, P0, Q1, f128s, 4096, 128, 128, PB, PB, 0, 32,  1.f, 1, nullptr, 0, sel, 1, 1);
  gemm(stream, Q0, Q1, f128c, 4096, 128, 128, PB, PB, 0, 32,  1.f, 0, nullptr, 0, sel, 1, 1);
  gemm(stream, Q0, P1, f128s, 4096, 128, 128, PB, PB, 0, 32, -1.f, 1, nullptr, 0, sel, 1, 1);
  // multiply by we
  scale_we<<<(unsigned)NB, 256, 0, stream>>>(P0, Q0, webuf, sel, 1);
  // inverse C: R = (R@Fc - I@Fs)/128 ; I = (I@Fc + R@Fs)/128
  gemm(stream, P1, P0, f128c, 4096, 128, 128, PB, PB, 0, 32,  1.f / 128.f, 0, nullptr, 0, sel, 1, 1);
  gemm(stream, P1, Q0, f128s, 4096, 128, 128, PB, PB, 0, 32, -1.f / 128.f, 1, nullptr, 0, sel, 1, 1);
  gemm(stream, Q1, Q0, f128c, 4096, 128, 128, PB, PB, 0, 32,  1.f / 128.f, 0, nullptr, 0, sel, 1, 1);
  gemm(stream, Q1, P0, f128s, 4096, 128, 128, PB, PB, 0, 32,  1.f / 128.f, 1, nullptr, 0, sel, 1, 1);
  // inverse W: R = (C@R - S@I)/64 ; I = (C@I + S@R)/64
  gemm(stream, P0, f64c, P1, 64, 128, 64, 8192, 0, 8192, 2048,  1.f / 64.f, 0, nullptr, 0, sel, 64, 1);
  gemm(stream, P0, f64s, Q1, 64, 128, 64, 8192, 0, 8192, 2048, -1.f / 64.f, 1, nullptr, 0, sel, 64, 1);
  gemm(stream, Q0, f64c, Q1, 64, 128, 64, 8192, 0, 8192, 2048,  1.f / 64.f, 0, nullptr, 0, sel, 64, 1);
  gemm(stream, Q0, f64s, P1, 64, 128, 64, 8192, 0, 8192, 2048,  1.f / 64.f, 1, nullptr, 0, sel, 64, 1);
  // inverse H, real part only -> xo
  gemm(stream, xo, f64c, P0, 64, 8192, 64, PB, 0, PB, 32,  1.f / 64.f, 0, nullptr, 0, sel, 1, 1);
  gemm(stream, xo, f64s, Q0, 64, 8192, 64, PB, 0, PB, 32, -1.f / 64.f, 1, nullptr, 0, sel, 1, 1);

  // ---------------- Haar branch (sel==2) ----------------
  haar_fwd<<<(unsigned)NB, 256, 0, stream>>>(x_, webuf, sel, P0);
  haar_inv<<<(unsigned)NB, 256, 0, stream>>>(P0, sel, xo);

  // ---------------- tail: LN * silu(z), final linear, transpose ----------------
  ln_silu<<<16384, 256, 0, stream>>>(xo, zs, ong, onb, x_);
  gemm(stream, P0, x_, olw, 131072, 128, 128, 0, 0, 0, 1, 1.f, 0, olb, 1,
       nullptr, 1, 0);
  transpose_out<<<(unsigned)NB, 256, 0, stream>>>(P0, (float*)d_out);
}
